// ShapeRenderer_35923106464315
// MI455X (gfx1250) — compile-verified, matrix operand order checked by reading, not run on hardware
//
#include <hip/hip_runtime.h>
#include <math.h>

#define NRAYS 2048
#define SMAX 128

typedef float     v8f  __attribute__((ext_vector_type(8)));
typedef _Float16  v16h __attribute__((ext_vector_type(16)));
typedef _Float16  v8h  __attribute__((ext_vector_type(8)));

// Padded layer geometry: 8 SDF layers then 4 color layers.
// Wt is stored transposed: [Opad][Kpad], f16, zero padded.
// Layer 7 output padded to 160 (not 144) so 2x2 WMMA blocking divides evenly.
static constexpr int    KPAD[12]  = {64,256,256,256,320,256,256,256, 160,256,256,256};
static constexpr int    OPAD[12]  = {256,256,256,256,256,256,256,160, 256,256,256,16};
static constexpr size_t WTOFF[12] = {0,16384,81920,147456,212992,294912,360448,425984,
                                     466944,507904,573440,638976};
#define WT_TOTAL 643072
#define BIAS_POOL (12*256)

// ---------------------------------------------------------------- math helpers
__device__ __forceinline__ float sigmoidf_(float x) { return 1.f / (1.f + expf(-x)); }
__device__ __forceinline__ float softplus100f(float x) {
  float t = 100.f * x;
  return (t > 20.f) ? x : log1pf(expf(t)) * 0.01f;
}

// posenc element c of a 64-wide padded input (x,y,z, then 6 freq blocks of sin3+cos3)
__device__ __forceinline__ float posenc_elem(const float* pt, int c) {
  if (c < 3)  return pt[c];
  if (c >= 39) return 0.f;
  int f = c - 3, fb = f / 6, w = f - fb * 6;
  float freq = (float)(1 << fb);
  if (w < 3) return sinf(freq * pt[w]);
  return cosf(freq * pt[w - 3]);
}
// d(posenc_c)/d(pt[a])
__device__ __forceinline__ float dposenc_elem(const float* pt, int c, int a) {
  if (c < 3)  return (c == a) ? 1.f : 0.f;
  if (c >= 39) return 0.f;
  int f = c - 3, fb = f / 6, w = f - fb * 6;
  float freq = (float)(1 << fb);
  if (w < 3)  return (w == a) ? freq * cosf(freq * pt[a]) : 0.f;
  return (w - 3 == a) ? -freq * sinf(freq * pt[a]) : 0.f;
}

// Load 16 f16 for one WMMA fragment lane: elems {p[0..7], p[16..23]} per the
// CDNA5 16-bit A/B VGPR layout (K = kb+{0..7} and kb+16+{0..7}).
__device__ __forceinline__ v16h ld16h(const _Float16* p) {
  v8h lo = *(const v8h*)(p);
  v8h hi = *(const v8h*)(p + 16);
  v16h r;
#pragma unroll
  for (int i = 0; i < 8; ++i) { r[i] = lo[i]; r[i + 8] = hi[i]; }
  return r;
}

__device__ __forceinline__ v8f wmma_f16(v16h a, v16h b, v8f c) {
  return __builtin_amdgcn_wmma_f32_16x16x32_f16(false, a, false, b, (short)0,
                                                c, false, false);
}

// ------------------------------------------------------------------ core GEMM
// 64-row LDS activation panel @ transposed global weights -> raw f16 panel.
// 2x2 register blocking: each wave owns a 32x32 macro-tile (4 independent
// accumulators) so the 4 WMMAs per K-step form independent chains (no F16
// WMMA->WMMA hazard NOPs) and each A/B fragment feeds two WMMAs.
// Requires Opad % 32 == 0.  Optional second K-range source s1 (skip concat)
// whose weights sit at K index K0.. in Wt.
__device__ __forceinline__ void wave_gemm64x2(
    const _Float16* __restrict__ Wt, int Kpad, int Opad,
    const _Float16* s0, int st0, int K0,
    const _Float16* s1, int st1, int K1,
    _Float16* gout, int gst)
{
  const int lane = threadIdx.x & 31;
  const int wave = threadIdx.x >> 5;
  const int nrm  = lane & 15;          // M for A / N for B / N for C
  const int hs8  = (lane >> 4) << 3;   // K (A,B) / M (C) offset for upper half-wave
  const int cmn  = Opad >> 5;          // macro cols
  const int nmacro = 2 * cmn;          // 64 rows = 2 macro rows
  for (int m = wave; m < nmacro; m += 8) {
    const int rtm = m / cmn;
    const int ctm = m - rtm * cmn;
    const int arow = rtm * 32 + nrm;
    const _Float16* __restrict__ b0p = Wt + (size_t)(ctm * 32 + nrm) * Kpad;
    const _Float16* __restrict__ b1p = b0p + (size_t)16 * Kpad;
    __builtin_prefetch(b0p, 0, 1);
    v8f c00, c01, c10, c11;
#pragma unroll
    for (int i = 0; i < 8; ++i) { c00[i]=0.f; c01[i]=0.f; c10[i]=0.f; c11[i]=0.f; }
    for (int k = 0; k < K0; k += 32) {
      v16h A0 = ld16h(s0 + arow * st0 + k + hs8);
      v16h A1 = ld16h(s0 + (arow + 16) * st0 + k + hs8);
      v16h B0 = ld16h(b0p + k + hs8);
      v16h B1 = ld16h(b1p + k + hs8);
      c00 = wmma_f16(A0, B0, c00);
      c01 = wmma_f16(A0, B1, c01);
      c10 = wmma_f16(A1, B0, c10);
      c11 = wmma_f16(A1, B1, c11);
    }
    for (int k = 0; k < K1; k += 32) {          // skip-connection K range
      v16h A0 = ld16h(s1 + arow * st1 + k + hs8);
      v16h A1 = ld16h(s1 + (arow + 16) * st1 + k + hs8);
      v16h B0 = ld16h(b0p + K0 + k + hs8);
      v16h B1 = ld16h(b1p + K0 + k + hs8);
      c00 = wmma_f16(A0, B0, c00);
      c01 = wmma_f16(A0, B1, c01);
      c10 = wmma_f16(A1, B0, c10);
      c11 = wmma_f16(A1, B1, c11);
    }
    // C layout: vgpr r holds row r (lanes 0-15) / r+8 (lanes 16-31)
    _Float16* g00 = gout + (size_t)(rtm * 32 + hs8) * gst + ctm * 32 + nrm;
#pragma unroll
    for (int r = 0; r < 8; ++r) {
      g00[r * gst]              = (_Float16)c00[r];
      g00[r * gst + 16]         = (_Float16)c01[r];
      g00[(r + 16) * gst]       = (_Float16)c10[r];
      g00[(r + 16) * gst + 16]  = (_Float16)c11[r];
    }
  }
}

// Simple single-tile path for tiny (16-column) output heads.
__device__ __forceinline__ void wave_gemm64(
    const _Float16* __restrict__ Wt, int Kpad, int Opad,
    const _Float16* s0, int st0, int K0,
    _Float16* gout, int gst)
{
  const int lane = threadIdx.x & 31;
  const int wave = threadIdx.x >> 5;
  const int nrm  = lane & 15;
  const int hs8  = (lane >> 4) << 3;
  const int ctn  = Opad >> 4;
  const int ntile = 4 * ctn;
  for (int t = wave; t < ntile; t += 8) {
    const int rt = t / ctn;
    const int ct = t - rt * ctn;
    v8f acc;
#pragma unroll
    for (int i = 0; i < 8; ++i) acc[i] = 0.f;
    const int arow = rt * 16 + nrm;
    const _Float16* __restrict__ brow = Wt + (size_t)(ct * 16 + nrm) * Kpad;
    for (int k = 0; k < K0; k += 32) {
      v16h a = ld16h(s0 + arow * st0 + k + hs8);
      v16h b = ld16h(brow + k + hs8);
      acc = wmma_f16(a, b, acc);
    }
    _Float16* gp = gout + (size_t)(rt * 16 + hs8) * gst + ct * 16 + nrm;
#pragma unroll
    for (int r = 0; r < 8; ++r) gp[r * gst] = (_Float16)acc[r];
  }
}

// elementwise passes over a 64x256 panel (raw g -> activated h)
__device__ __forceinline__ void act_sp100(const _Float16* g, _Float16* h, const float* b) {
  for (int e = threadIdx.x; e < 64 * 256; e += 256) {
    int c = e & 255;
    h[e] = (_Float16)softplus100f((float)g[e] + b[c]);
  }
}
__device__ __forceinline__ void act_relu(const _Float16* g, _Float16* h, const float* b) {
  for (int e = threadIdx.x; e < 64 * 256; e += 256) {
    int c = e & 255;
    h[e] = (_Float16)fmaxf((float)g[e] + b[c], 0.f);
  }
}
// JVP panel: rows r%4==0 are values (softplus), others tangents (g * sigmoid(100*pre_val))
__device__ __forceinline__ void act_jvp(const _Float16* g, _Float16* h, const float* b) {
  for (int e = threadIdx.x; e < 64 * 256; e += 256) {
    int r = e >> 8, c = e & 255;
    int vr = r & ~3;
    float gv = (float)g[(vr << 8) + c] + b[c];
    float out = ((r & 3) == 0) ? softplus100f(gv)
                               : (float)g[e] * sigmoidf_(100.f * gv);
    h[e] = (_Float16)out;
  }
}

// ------------------------------------------------------------------- kernels
__global__ void cvt_kernel(const float* __restrict__ W, const float* __restrict__ B,
                           _Float16* __restrict__ Wt, float* __restrict__ Bt,
                           int realIn, int realOut, int Kpad, int Opad, float scale)
{
  int idx = blockIdx.x * blockDim.x + threadIdx.x;
  int total = Kpad * Opad;
  if (idx < total) {
    int o = idx / Kpad;
    int k = idx - o * Kpad;
    float v = (k < realIn && o < realOut) ? W[(size_t)k * realOut + o] * scale : 0.f;
    Wt[(size_t)o * Kpad + k] = (_Float16)v;
  }
  if (idx < Opad) Bt[idx] = (idx < realOut) ? B[idx] : 0.f;
}

__global__ void ray_init_kernel(const float* __restrict__ ro, const float* __restrict__ rd,
                                float* __restrict__ zbuf)
{
  int ray = blockIdx.x * blockDim.x + threadIdx.x;
  if (ray >= NRAYS) return;
  float ox = ro[3*ray], oy = ro[3*ray+1], oz = ro[3*ray+2];
  float dx = rd[3*ray], dy = rd[3*ray+1], dz = rd[3*ray+2];
  float mid  = -(ox*dx + oy*dy + oz*dz);
  float nearv = fmaxf(mid - 1.f, 0.05f);
  float farv  = mid + 1.f;
  for (int i = 0; i < 64; ++i)
    zbuf[ray * SMAX + i] = nearv + (farv - nearv) * ((float)i / 63.f);
}

// SDF forward only (for up-sampling passes); 64 points per workgroup.
__global__ __launch_bounds__(256)
void sdf_eval_kernel(const float* __restrict__ ro, const float* __restrict__ rd,
                     const float* __restrict__ zbuf,
                     const _Float16* __restrict__ wt, const float* __restrict__ bias,
                     float* __restrict__ sdf_out, int S)
{
  __shared__ alignas(16) _Float16 sI[64 * 64];
  __shared__ alignas(16) _Float16 sA[64 * 256];
  __shared__ alignas(16) _Float16 sG[64 * 256];
  const int base = blockIdx.x * 64;
  for (int e = threadIdx.x; e < 64 * 64; e += 256) {
    int r = e >> 6, c = e & 63;
    int p = base + r;
    int ray = p / S, s = p - ray * S;
    float z = zbuf[ray * SMAX + s];
    float pt[3];
#pragma unroll
    for (int j = 0; j < 3; ++j) pt[j] = ro[3*ray+j] + rd[3*ray+j] * z;
    sI[e] = (_Float16)posenc_elem(pt, c);
  }
  __syncthreads();
  wave_gemm64x2(wt + WTOFF[0], 64, 256, sI, 64, 64, nullptr, 0, 0, sG, 256);
  __syncthreads(); act_sp100(sG, sA, bias + 0*256); __syncthreads();
  wave_gemm64x2(wt + WTOFF[1], 256, 256, sA, 256, 256, nullptr, 0, 0, sG, 256);
  __syncthreads(); act_sp100(sG, sA, bias + 1*256); __syncthreads();
  wave_gemm64x2(wt + WTOFF[2], 256, 256, sA, 256, 256, nullptr, 0, 0, sG, 256);
  __syncthreads(); act_sp100(sG, sA, bias + 2*256); __syncthreads();
  wave_gemm64x2(wt + WTOFF[3], 256, 256, sA, 256, 256, nullptr, 0, 0, sG, 256);
  __syncthreads(); act_sp100(sG, sA, bias + 3*256); __syncthreads();
  wave_gemm64x2(wt + WTOFF[4], 320, 256, sA, 256, 256, sI, 64, 64, sG, 256);  // skip concat
  __syncthreads(); act_sp100(sG, sA, bias + 4*256); __syncthreads();
  wave_gemm64x2(wt + WTOFF[5], 256, 256, sA, 256, 256, nullptr, 0, 0, sG, 256);
  __syncthreads(); act_sp100(sG, sA, bias + 5*256); __syncthreads();
  wave_gemm64x2(wt + WTOFF[6], 256, 256, sA, 256, 256, nullptr, 0, 0, sG, 256);
  __syncthreads(); act_sp100(sG, sA, bias + 6*256); __syncthreads();
  wave_gemm64(wt + WTOFF[7], 256, 16, sA, 256, 256, sG, 256);  // only sdf col tile
  __syncthreads();
  for (int r = threadIdx.x; r < 64; r += 256)
    sdf_out[base + r] = (float)sG[r * 256] + bias[7*256];
}

// Per-ray NeuS up-sampling (weights -> sample_pdf -> sorted merge), 1 thread/ray.
__global__ void upsample_kernel(float* __restrict__ zbuf, const float* __restrict__ sdfb,
                                int S, float inv_s)
{
  int ray = blockIdx.x * blockDim.x + threadIdx.x;
  if (ray >= NRAYS) return;
  float z[SMAX], sd[SMAX], raw[SMAX], wgt[SMAX], cdf[SMAX];
  for (int i = 0; i < S; ++i) { z[i] = zbuf[ray * SMAX + i]; sd[i] = sdfb[ray * S + i]; }
  for (int i = 0; i < S - 1; ++i)
    raw[i] = (sd[i+1] - sd[i]) / (z[i+1] - z[i] + 1e-5f);
  float T = 1.f, wsum = 0.f;
  for (int i = 0; i < S - 1; ++i) {
    float pc = (i == 0) ? 0.f : raw[i-1];
    float cv = fminf(raw[i], pc);
    cv = fmaxf(fminf(cv, 0.f), -1000.f);
    float m = 0.5f * (sd[i] + sd[i+1]);
    float dist = z[i+1] - z[i];
    float pe = m - cv * dist * 0.5f, ne = m + cv * dist * 0.5f;
    float pcdf = sigmoidf_(pe * inv_s), ncdf = sigmoidf_(ne * inv_s);
    float a = (pcdf - ncdf + 1e-5f) / (pcdf + 1e-5f);
    float w = a * T;
    T *= (1.f - a + 1e-7f);
    wgt[i] = w + 1e-5f;
    wsum += wgt[i];
  }
  cdf[0] = 0.f;
  for (int i = 1; i < S; ++i) cdf[i] = cdf[i-1] + wgt[i-1] / wsum;
  float nz[16];
  int pos = 0;
  for (int j = 0; j < 16; ++j) {
    float u = (j + 0.5f) / 16.f;
    while (pos < S && cdf[pos] <= u) ++pos;
    int below = pos - 1; below = below < 0 ? 0 : (below > S-1 ? S-1 : below);
    int above = pos > S-1 ? S-1 : pos;
    float cb = cdf[below], ca = cdf[above];
    float den = (ca - cb < 1e-5f) ? 1.f : (ca - cb);
    float t = (u - cb) / den;
    nz[j] = z[below] + t * (z[above] - z[below]);
  }
  int i = 0, j = 0;
  for (int k = 0; k < S + 16; ++k) {
    float v;
    if (j >= 16 || (i < S && z[i] <= nz[j])) v = z[i++]; else v = nz[j++];
    zbuf[ray * SMAX + k] = v;
  }
}

__global__ void midz_kernel(const float* __restrict__ ro, const float* __restrict__ rd,
                            const float* __restrict__ zbuf,
                            float* __restrict__ midz, float* __restrict__ dists)
{
  int ray = blockIdx.x * blockDim.x + threadIdx.x;
  if (ray >= NRAYS) return;
  float ox = ro[3*ray], oy = ro[3*ray+1], oz = ro[3*ray+2];
  float dx = rd[3*ray], dy = rd[3*ray+1], dz = rd[3*ray+2];
  float mid = -(ox*dx + oy*dy + oz*dz);
  float nearv = fmaxf(mid - 1.f, 0.05f);
  float farv  = mid + 1.f;
  float sdist = (farv - nearv) / 64.f;   // (far-near)/N_SAMPLES
  for (int i = 0; i < SMAX; ++i) {
    float zi = zbuf[ray * SMAX + i];
    float d = (i < SMAX - 1) ? (zbuf[ray * SMAX + i + 1] - zi) : sdist;
    dists[ray * SMAX + i] = d;
    midz [ray * SMAX + i] = zi + 0.5f * d;
  }
}

// SDF forward + forward-mode JVP gradient; 16 points (= 64 rows) per workgroup.
__global__ __launch_bounds__(256)
void sdf_full_kernel(const float* __restrict__ ro, const float* __restrict__ rd,
                     const float* __restrict__ midz,
                     const _Float16* __restrict__ wt, const float* __restrict__ bias,
                     float* __restrict__ sdf_out, _Float16* __restrict__ feat,
                     float* __restrict__ grad)
{
  __shared__ alignas(16) _Float16 sI[64 * 64];
  __shared__ alignas(16) _Float16 sA[64 * 256];
  __shared__ alignas(16) _Float16 sG[64 * 256];
  const int base = blockIdx.x * 16;   // points
  for (int e = threadIdx.x; e < 64 * 64; e += 256) {
    int r = e >> 6, c = e & 63;
    int p = base + (r >> 2), sub = r & 3;
    int ray = p >> 7;
    float z = midz[p];
    float pt[3];
#pragma unroll
    for (int j = 0; j < 3; ++j) pt[j] = ro[3*ray+j] + rd[3*ray+j] * z;
    sI[e] = (_Float16)((sub == 0) ? posenc_elem(pt, c) : dposenc_elem(pt, c, sub - 1));
  }
  __syncthreads();
  wave_gemm64x2(wt + WTOFF[0], 64, 256, sI, 64, 64, nullptr, 0, 0, sG, 256);
  __syncthreads(); act_jvp(sG, sA, bias + 0*256); __syncthreads();
  wave_gemm64x2(wt + WTOFF[1], 256, 256, sA, 256, 256, nullptr, 0, 0, sG, 256);
  __syncthreads(); act_jvp(sG, sA, bias + 1*256); __syncthreads();
  wave_gemm64x2(wt + WTOFF[2], 256, 256, sA, 256, 256, nullptr, 0, 0, sG, 256);
  __syncthreads(); act_jvp(sG, sA, bias + 2*256); __syncthreads();
  wave_gemm64x2(wt + WTOFF[3], 256, 256, sA, 256, 256, nullptr, 0, 0, sG, 256);
  __syncthreads(); act_jvp(sG, sA, bias + 3*256); __syncthreads();
  wave_gemm64x2(wt + WTOFF[4], 320, 256, sA, 256, 256, sI, 64, 64, sG, 256);
  __syncthreads(); act_jvp(sG, sA, bias + 4*256); __syncthreads();
  wave_gemm64x2(wt + WTOFF[5], 256, 256, sA, 256, 256, nullptr, 0, 0, sG, 256);
  __syncthreads(); act_jvp(sG, sA, bias + 5*256); __syncthreads();
  wave_gemm64x2(wt + WTOFF[6], 256, 256, sA, 256, 256, nullptr, 0, 0, sG, 256);
  __syncthreads(); act_jvp(sG, sA, bias + 6*256); __syncthreads();
  wave_gemm64x2(wt + WTOFF[7], 256, 160, sA, 256, 256, nullptr, 0, 0, sG, 256);
  __syncthreads();
  for (int e = threadIdx.x; e < 64 * 144; e += 256) {
    int r = e / 144, c = e - 144 * r;
    int p = base + (r >> 2), sub = r & 3;
    float g = (float)sG[r * 256 + c];
    if (sub == 0) {
      if (c == 0)       sdf_out[p] = g + bias[7*256];
      else if (c < 129) feat[(size_t)p * 128 + (c - 1)] = (_Float16)(g + bias[7*256 + c]);
    } else if (c == 0) {
      grad[p * 3 + sub - 1] = g;     // d sdf / d pt[sub-1]
    }
  }
}

// Color MLP; 64 points per workgroup.
__global__ __launch_bounds__(256)
void color_kernel(const float* __restrict__ ro, const float* __restrict__ rd,
                  const float* __restrict__ midz,
                  const _Float16* __restrict__ feat, const float* __restrict__ grad,
                  const _Float16* __restrict__ wt, const float* __restrict__ bias,
                  float* __restrict__ rgb)
{
  __shared__ alignas(16) _Float16 sI[64 * 160];
  __shared__ alignas(16) _Float16 sA[64 * 256];
  __shared__ alignas(16) _Float16 sG[64 * 256];
  const int base = blockIdx.x * 64;
  for (int e = threadIdx.x; e < 64 * 160; e += 256) {
    int r = e / 160, c = e - 160 * r;
    int p = base + r;
    int ray = p >> 7;
    float v;
    if (c < 3) {
      v = ro[3*ray+c] + rd[3*ray+c] * midz[p];
    } else if (c < 6) {
      v = rd[3*ray + c - 3];
    } else if (c < 9) {
      float gx = grad[3*p], gy = grad[3*p+1], gz = grad[3*p+2];
      float n = sqrtf(gx*gx + gy*gy + gz*gz) + 1e-6f;
      v = grad[3*p + c - 6] / n;
    } else if (c < 137) {
      v = (float)feat[(size_t)p * 128 + (c - 9)];
    } else v = 0.f;
    sI[e] = (_Float16)v;
  }
  __syncthreads();
  wave_gemm64x2(wt + WTOFF[8], 160, 256, sI, 160, 160, nullptr, 0, 0, sG, 256);
  __syncthreads(); act_relu(sG, sA, bias + 8*256); __syncthreads();
  wave_gemm64x2(wt + WTOFF[9], 256, 256, sA, 256, 256, nullptr, 0, 0, sG, 256);
  __syncthreads(); act_relu(sG, sA, bias + 9*256); __syncthreads();
  wave_gemm64x2(wt + WTOFF[10], 256, 256, sA, 256, 256, nullptr, 0, 0, sG, 256);
  __syncthreads(); act_relu(sG, sA, bias + 10*256); __syncthreads();
  wave_gemm64(wt + WTOFF[11], 256, 16, sA, 256, 256, sG, 256);
  __syncthreads();
  for (int e = threadIdx.x; e < 64 * 3; e += 256) {
    int r = e / 3, c = e - 3 * r;
    rgb[(size_t)(base + r) * 3 + c] = sigmoidf_((float)sG[r * 256 + c] + bias[11*256 + c]);
  }
}

// Final NeuS compositing; 1 thread/ray.
__global__ void composite_kernel(const float* __restrict__ rd, const float* __restrict__ dists,
                                 const float* __restrict__ sdfb, const float* __restrict__ grad,
                                 const float* __restrict__ rgb, const float* __restrict__ var,
                                 float* __restrict__ out)
{
  int ray = blockIdx.x * blockDim.x + threadIdx.x;
  if (ray >= NRAYS) return;
  float inv_s = expf(10.f * var[0]);
  inv_s = fminf(fmaxf(inv_s, 1e-6f), 1e6f);
  float dx = rd[3*ray], dy = rd[3*ray+1], dz = rd[3*ray+2];
  float T = 1.f, acc = 0.f, c0 = 0.f, c1 = 0.f, c2 = 0.f;
  for (int i = 0; i < SMAX; ++i) {
    int p = ray * SMAX + i;
    float s = sdfb[p];
    float tc = dx*grad[3*p] + dy*grad[3*p+1] + dz*grad[3*p+2];
    float ic = fminf(tc, 0.f);                 // -relu(-true_cos)
    float d = dists[p];
    float ep = s - ic * d * 0.5f, en = s + ic * d * 0.5f;
    float pc = sigmoidf_(ep * inv_s), nc = sigmoidf_(en * inv_s);
    float a = (pc - nc + 1e-5f) / (pc + 1e-5f);
    a = fminf(fmaxf(a, 0.f), 1.f);
    float w = a * T;
    T *= (1.f - a + 1e-7f);
    c0 += w * rgb[3*p]; c1 += w * rgb[3*p+1]; c2 += w * rgb[3*p+2];
    acc += w;
  }
  out[3*ray]   = c0 + (1.f - acc);
  out[3*ray+1] = c1 + (1.f - acc);
  out[3*ray+2] = c2 + (1.f - acc);
}

// ------------------------------------------------------------------- launcher
extern "C" void kernel_launch(void* const* d_in, const int* in_sizes, int n_in,
                              void* d_out, int out_size, void* d_ws, size_t ws_size,
                              hipStream_t stream)
{
  (void)in_sizes; (void)n_in; (void)out_size; (void)ws_size;
  const float* ro = (const float*)d_in[0];
  const float* rd = (const float*)d_in[1];
  const float* W[12]; const float* B[12];
  for (int i = 0; i < 8; ++i) { W[i]   = (const float*)d_in[2 + 2*i];  B[i]   = (const float*)d_in[3 + 2*i]; }
  for (int i = 0; i < 4; ++i) { W[8+i] = (const float*)d_in[18 + 2*i]; B[8+i] = (const float*)d_in[19 + 2*i]; }
  const float* var = (const float*)d_in[26];

  char* ws = (char*)d_ws;
  size_t off = 0;
  auto take = [&](size_t bytes) -> void* {
    void* p = ws + off;
    off += (bytes + 255) & ~(size_t)255;
    return p;
  };
  _Float16* wt   = (_Float16*)take((size_t)WT_TOTAL * sizeof(_Float16));
  float*    bias = (float*)   take((size_t)BIAS_POOL * sizeof(float));
  float*    zbuf = (float*)   take((size_t)NRAYS * SMAX * sizeof(float));
  float*    sdfb = (float*)   take((size_t)NRAYS * SMAX * sizeof(float));
  float*    midz = (float*)   take((size_t)NRAYS * SMAX * sizeof(float));
  float*    dsts = (float*)   take((size_t)NRAYS * SMAX * sizeof(float));
  float*    grad = (float*)   take((size_t)NRAYS * SMAX * 3 * sizeof(float));
  float*    rgb  = (float*)   take((size_t)NRAYS * SMAX * 3 * sizeof(float));
  _Float16* feat = (_Float16*)take((size_t)NRAYS * SMAX * 128 * sizeof(_Float16));

  static const int realIn[12]  = {39,256,256,256,295,256,256,256, 137,256,256,256};
  static const int realOut[12] = {256,256,256,256,256,256,256,129, 256,256,256,3};
  for (int i = 0; i < 12; ++i) {
    int total = KPAD[i] * OPAD[i];
    float scale = (i == 4) ? 0.70710678118654752f : 1.f;   // fold concat /sqrt(2) into W4
    cvt_kernel<<<(total + 255) / 256, 256, 0, stream>>>(
        W[i], B[i], wt + WTOFF[i], bias + i * 256,
        realIn[i], realOut[i], KPAD[i], OPAD[i], scale);
  }

  ray_init_kernel<<<(NRAYS + 255) / 256, 256, 0, stream>>>(ro, rd, zbuf);

  for (int step = 0; step < 4; ++step) {
    int S = 64 + 16 * step;
    int P = NRAYS * S;
    sdf_eval_kernel<<<P / 64, 256, 0, stream>>>(ro, rd, zbuf, wt, bias, sdfb, S);
    float inv_s = 64.f * (float)(1 << step);
    upsample_kernel<<<(NRAYS + 255) / 256, 256, 0, stream>>>(zbuf, sdfb, S, inv_s);
  }

  midz_kernel<<<(NRAYS + 255) / 256, 256, 0, stream>>>(ro, rd, zbuf, midz, dsts);
  sdf_full_kernel<<<(NRAYS * SMAX) / 16, 256, 0, stream>>>(ro, rd, midz, wt, bias, sdfb, feat, grad);
  color_kernel<<<(NRAYS * SMAX) / 64, 256, 0, stream>>>(ro, rd, midz, feat, grad, wt, bias, rgb);
  composite_kernel<<<(NRAYS + 255) / 256, 256, 0, stream>>>(rd, dsts, sdfb, grad, rgb, var, (float*)d_out);
}
